// BHAttention_84018150244598
// MI455X (gfx1250) — matrix operand order repaired and checked by reading, not compiled, not run on hardware
//
#include <hip/hip_runtime.h>
#include <hip/hip_bf16.h>

typedef __attribute__((ext_vector_type(16))) _Float16 v16h;
typedef __attribute__((ext_vector_type(8)))  _Float16 v8h;
typedef __attribute__((ext_vector_type(8)))  float    v8f;

#define NB    8      // batch
#define CDIM  512    // input channels
#define HRES  128    // input spatial
#define DD    128    // inner dim
#define WSZ   32     // window side
#define NTOK  1024   // tokens
#define NHEAD 4
#define HDIM  32

// ---------------------------------------------------------------------------
// WMMA helpers (CDNA5 16x16x32 f16 -> f32, per CDNA5 ISA layouts)
// ---------------------------------------------------------------------------
__device__ __forceinline__ v8f wmma16(v16h a, v16h b, v8f c) {
    return __builtin_amdgcn_wmma_f32_16x16x32_f16(
        /*neg_a=*/false, a, /*neg_b=*/false, b,
        /*c_mod=*/(short)0, c, /*reuse_a=*/false, /*reuse_b=*/false);
}

__device__ __forceinline__ v16h pack16(v8h lo, v8h hi) {
    v16h r;
#pragma unroll
    for (int i = 0; i < 8; ++i) { r[i] = lo[i]; r[8 + i] = hi[i]; }
    return r;
}

// A-matrix 16x32 (MxK), row-major storage with leading dim `ld` (halves).
// lane<16: K {0..7,16..23}; lane>=16: K {8..15,24..31}; row = lane&15.
__device__ __forceinline__ v16h load_a_tile(const _Float16* p, int ld, int lane) {
    const _Float16* base = p + (lane & 15) * ld + (lane >> 4) * 8;
    v8h lo = *(const v8h*)base;
    v8h hi = *(const v8h*)(base + 16);
    return pack16(lo, hi);
}

// B-matrix 32x16 (KxN), element (k,n) stored at p[n*ld + k] (K contiguous).
// lane<16: N=lane, K {0..7,16..23}; lane>=16: N=lane-16, K {8..15,24..31}.
__device__ __forceinline__ v16h load_b_tile(const _Float16* p, int ld, int lane) {
    const _Float16* base = p + (lane & 15) * ld + (lane >> 4) * 8;
    v8h lo = *(const v8h*)base;
    v8h hi = *(const v8h*)(base + 16);
    return pack16(lo, hi);
}

// ---------------------------------------------------------------------------
// 0) weight prep: transpose + f16-convert the dense weights
// ---------------------------------------------------------------------------
__global__ __launch_bounds__(256) void prep_kernel(
    const float* __restrict__ qw, const float* __restrict__ kvw,
    const float* __restrict__ pw, const float* __restrict__ cow,
    _Float16* __restrict__ qwT, _Float16* __restrict__ kvT,
    _Float16* __restrict__ pwT, _Float16* __restrict__ cowT) {
    int i = blockIdx.x * 256 + threadIdx.x;            // 65536 threads
    if (i < 128 * 128) {
        int k = i >> 7, n = i & 127;
        qwT[n * 128 + k] = (_Float16)qw[i];
        pwT[n * 128 + k] = (_Float16)pw[i];
    }
    if (i < 128 * 256) {
        int k = i >> 8, n = i & 255;
        kvT[n * 128 + k] = (_Float16)kvw[i];
    }
    cowT[i] = (_Float16)cow[i];                        // co_w is already [out][in]
}

// ---------------------------------------------------------------------------
// 1) fused front-end: grouped conv4x4/s4 + max/avg pool + 1x1 gconv + BN + relu6
//    reads x exactly once (256 MB), emits tq/tm/ta f16 token maps [8192][128]
// ---------------------------------------------------------------------------
struct FrontArgs {
    const float *x;
    const float *le_w, *le_b, *le_g, *le_beta, *le_m, *le_v;
    const float *mx_w, *mx_b, *mx_g, *mx_beta, *mx_m, *mx_v;
    const float *av_w, *av_b, *av_g, *av_beta, *av_m, *av_v;
    _Float16 *tq, *tm, *ta;
};

__device__ __forceinline__ float bn_relu6(float y, float g, float be, float m, float v) {
    float s = g * __frsqrt_rn(v + 1e-5f);
    float r = (y - m) * s + be;
    return fminf(fmaxf(r, 0.0f), 6.0f);
}

__global__ __launch_bounds__(256) void frontend_kernel(FrontArgs a) {
    int t   = threadIdx.x;
    int blk = blockIdx.x;                // b*512 + d*4 + seg
    int seg = blk & 3;
    int d   = (blk >> 2) & 127;
    int b   = blk >> 9;
    int oy  = seg * 8 + (t >> 5);
    int ox  = t & 31;

    const float* xb = a.x + (((b * CDIM + 4 * d) * HRES) + 4 * oy) * HRES + 4 * ox;
    float conv = 0.0f, mxv = 0.0f, avv = 0.0f;
#pragma unroll
    for (int c = 0; c < 4; ++c) {
        const float* xc = xb + c * (HRES * HRES);
        float cmax = -1e30f, csum = 0.0f;
#pragma unroll
        for (int ky = 0; ky < 4; ++ky) {
            float4 v = *(const float4*)(xc + ky * HRES);
            const float* lw = a.le_w + ((d * 4 + c) * 4 + ky) * 4;
            conv += v.x * lw[0] + v.y * lw[1] + v.z * lw[2] + v.w * lw[3];
            cmax = fmaxf(cmax, fmaxf(fmaxf(v.x, v.y), fmaxf(v.z, v.w)));
            csum += v.x + v.y + v.z + v.w;
        }
        mxv += a.mx_w[d * 4 + c] * cmax;
        avv += a.av_w[d * 4 + c] * (csum * (1.0f / 16.0f));
    }
    float q = bn_relu6(conv + a.le_b[d], a.le_g[d], a.le_beta[d], a.le_m[d], a.le_v[d]);
    float m = bn_relu6(mxv  + a.mx_b[d], a.mx_g[d], a.mx_beta[d], a.mx_m[d], a.mx_v[d]);
    float v = bn_relu6(avv  + a.av_b[d], a.av_g[d], a.av_beta[d], a.av_m[d], a.av_v[d]);

    int n   = oy * 32 + ox;
    int idx = (b * NTOK + n) * DD + d;
    a.tq[idx] = (_Float16)q;
    a.tm[idx] = (_Float16)m;
    a.ta[idx] = (_Float16)v;
}

// ---------------------------------------------------------------------------
// 2) WMMA GEMM: C[8192 x NOUT] = A[8192 x 128] * W (WT stored [nout][128]).
//    Each wave computes one 16-row M tile x two adjacent 16-col N tiles,
//    sharing the A fragment (8 WMMAs per wave). Templated epilogue ->
//    straight-line stores, no per-element scalar branching.
// ---------------------------------------------------------------------------
template <bool OUT_F16, bool HAS_BIAS>
__global__ __launch_bounds__(128) void gemm_wmma(
    const _Float16* __restrict__ A, const _Float16* __restrict__ WT,
    const float* __restrict__ bias, float bias_scale, float scale,
    int nout, void* __restrict__ outp) {
    int lane    = threadIdx.x & 31;
    int wave    = threadIdx.x >> 5;
    int npairs  = nout >> 5;                 // pairs of 16-wide N tiles
    int pair    = blockIdx.x * 4 + wave;
    int mt      = pair / npairs;
    int np      = pair - mt * npairs;
    int n0      = np * 32;                   // first column of the pair

    const _Float16* Ap = A + (mt * 16) * DD;
    const _Float16* W0 = WT + (n0     ) * DD;
    const _Float16* W1 = WT + (n0 + 16) * DD;
    v8f acc0 = {}, acc1 = {};
#pragma unroll
    for (int kc = 0; kc < 4; ++kc) {
        v16h a  = load_a_tile(Ap + kc * 32, DD, lane);
        v16h b0 = load_b_tile(W0 + kc * 32, DD, lane);
        v16h b1 = load_b_tile(W1 + kc * 32, DD, lane);
        acc0 = wmma16(a, b0, acc0);
        acc1 = wmma16(a, b1, acc1);
    }
    int   c0  = n0 + (lane & 15);
    int   c1  = c0 + 16;
    float bv0 = HAS_BIAS ? bias[c0] * bias_scale : 0.0f;
    float bv1 = HAS_BIAS ? bias[c1] * bias_scale : 0.0f;
#pragma unroll
    for (int e = 0; e < 8; ++e) {
        int   row = mt * 16 + e + (lane >> 4) * 8;
        float v0  = acc0[e] * scale + bv0;
        float v1  = acc1[e] * scale + bv1;
        if (OUT_F16) {
            ((_Float16*)outp)[row * nout + c0] = (_Float16)v0;
            ((_Float16*)outp)[row * nout + c1] = (_Float16)v1;
        } else {
            ((float*)outp)[row * nout + c0] = v0;
            ((float*)outp)[row * nout + c1] = v1;
        }
    }
}

// ---------------------------------------------------------------------------
// 3) transpose V halves of the kv outputs to [b][h][hd][token] (f16)
// ---------------------------------------------------------------------------
__global__ __launch_bounds__(256) void vt_kernel(
    const _Float16* __restrict__ kv1, const _Float16* __restrict__ kv2,
    _Float16* __restrict__ vt1, _Float16* __restrict__ vt2) {
    int i   = blockIdx.x * 256 + threadIdx.x;   // 8192*128
    int col = i & 127;                          // h*32 + dd
    int tg  = i >> 7;                           // b*1024 + tok
    int b   = tg >> 10, tok = tg & 1023;
    int h   = col >> 5, dd = col & 31;
    int dst = (((b * NHEAD + h) * HDIM + dd) << 10) + tok;
    vt1[dst] = kv1[tg * 256 + 128 + col];
    vt2[dst] = kv2[tg * 256 + 128 + col];
}

// ---------------------------------------------------------------------------
// 4) dual-branch windowed attention, two-pass softmax, WMMA everywhere.
//    Pass 1: S via WMMA, per-lane running max only (no cross-lane ops in loop).
//    One xor-reduction per branch. Pass 2: recompute S (WMMA is cheap),
//    P = exp(S - m), per-lane partial row sums, P*V via WMMA through LDS.
// ---------------------------------------------------------------------------
__global__ __launch_bounds__(128) void attn_kernel(
    const _Float16* __restrict__ qf,
    const _Float16* __restrict__ kv1, const _Float16* __restrict__ kv2,
    const _Float16* __restrict__ vt1, const _Float16* __restrict__ vt2,
    const float* __restrict__ rpb, _Float16* __restrict__ obuf) {
    __shared__ __align__(16) _Float16 plds[4][16][32];   // wave-private P tiles

    int lane = threadIdx.x & 31;
    int wave = threadIdx.x >> 5;
    int half = lane >> 4;
    int ln   = lane & 15;
    int h    = blockIdx.y;
    int b    = blockIdx.z;
    int row0 = blockIdx.x * 64 + wave * 16;    // query-token tile base

    // Q tile (A layout), scale already folded into qf
    v16h aq = load_a_tile(qf + (b * NTOK + row0) * DD + h * HDIM, DD, lane);

    int qy[8], qx[8];
#pragma unroll
    for (int e = 0; e < 8; ++e) {
        int t = row0 + e + half * 8;
        qy[e] = t >> 5; qx[e] = t & 31;
    }

    v8f os0 = {}, os1 = {};
    for (int br = 0; br < 2; ++br) {
        const _Float16* kv = br ? kv2 : kv1;
        const _Float16* vt = br ? vt2 : vt1;
        const _Float16* kbase = kv + (b * NTOK) * 256 + h * HDIM;
        const _Float16* vbase = vt + ((b * NHEAD + h) * HDIM) * NTOK;

        // ---- pass 1: exact row max, per-lane accumulation only ----
        float m[8];
#pragma unroll
        for (int e = 0; e < 8; ++e) m[e] = -1e30f;

        for (int j = 0; j < 32; ++j) {
            int tok0 = j * 32;
            const _Float16* kb = kbase + tok0 * 256;
            v16h bk0 = load_b_tile(kb, 256, lane);
            v16h bk1 = load_b_tile(kb + 16 * 256, 256, lane);
            v8f  z   = {};
            v8f  s0  = wmma16(aq, bk0, z);
            v8f  s1  = wmma16(aq, bk1, z);
            int kc0 = tok0 + ln, kc1 = kc0 + 16;
            int ky0 = kc0 >> 5, kx0 = kc0 & 31;
            int ky1 = kc1 >> 5, kx1 = kc1 & 31;
#pragma unroll
            for (int e = 0; e < 8; ++e) {
                int i0 = (qy[e] - ky0 + 31) * 63 + (qx[e] - kx0 + 31);
                int i1 = (qy[e] - ky1 + 31) * 63 + (qx[e] - kx1 + 31);
                float a0 = s0[e] + rpb[i0 * NHEAD + h];
                float a1 = s1[e] + rpb[i1 * NHEAD + h];
                m[e] = fmaxf(m[e], fmaxf(a0, a1));
            }
        }
        // single cross-lane max reduction per row (16-lane halves)
#pragma unroll
        for (int e = 0; e < 8; ++e) {
            float cm = m[e];
            cm = fmaxf(cm, __shfl_xor(cm, 1));
            cm = fmaxf(cm, __shfl_xor(cm, 2));
            cm = fmaxf(cm, __shfl_xor(cm, 4));
            cm = fmaxf(cm, __shfl_xor(cm, 8));
            m[e] = cm;
        }

        // ---- pass 2: P = exp(S - m), row sums (per-lane), O += P*V ----
        float ls[8];
#pragma unroll
        for (int e = 0; e < 8; ++e) ls[e] = 0.0f;
        v8f o0 = {}, o1 = {};

        for (int j = 0; j < 32; ++j) {
            int tok0 = j * 32;
            const _Float16* kb = kbase + tok0 * 256;
            __builtin_prefetch(kb + 32 * 256, 0, 0);
            v16h bk0 = load_b_tile(kb, 256, lane);
            v16h bk1 = load_b_tile(kb + 16 * 256, 256, lane);
            v8f  z   = {};
            v8f  s0  = wmma16(aq, bk0, z);
            v8f  s1  = wmma16(aq, bk1, z);
            int kc0 = tok0 + ln, kc1 = kc0 + 16;
            int ky0 = kc0 >> 5, kx0 = kc0 & 31;
            int ky1 = kc1 >> 5, kx1 = kc1 & 31;
#pragma unroll
            for (int e = 0; e < 8; ++e) {
                int i0 = (qy[e] - ky0 + 31) * 63 + (qx[e] - kx0 + 31);
                int i1 = (qy[e] - ky1 + 31) * 63 + (qx[e] - kx1 + 31);
                float a0 = s0[e] + rpb[i0 * NHEAD + h];
                float a1 = s1[e] + rpb[i1 * NHEAD + h];
                float p0 = __expf(a0 - m[e]);     // <= 1, f16-safe
                float p1 = __expf(a1 - m[e]);
                ls[e] += p0 + p1;
                int pr = e + half * 8;
                plds[wave][pr][ln]      = (_Float16)p0;
                plds[wave][pr][16 + ln] = (_Float16)p1;
            }
            // wave-private LDS: order ds_store -> ds_load (CDNA5 split counter)
            asm volatile("s_wait_dscnt 0" ::: "memory");
            v16h ap = load_a_tile(&plds[wave][0][0], 32, lane);

            const _Float16* vb = vbase + tok0;
            v16h bv0 = load_b_tile(vb, NTOK, lane);
            v16h bv1 = load_b_tile(vb + 16 * NTOK, NTOK, lane);
            o0 = wmma16(ap, bv0, o0);
            o1 = wmma16(ap, bv1, o1);
        }
        // single cross-lane sum reduction per row, then normalize+accumulate
#pragma unroll
        for (int e = 0; e < 8; ++e) {
            float rs = ls[e];
            rs += __shfl_xor(rs, 1);
            rs += __shfl_xor(rs, 2);
            rs += __shfl_xor(rs, 4);
            rs += __shfl_xor(rs, 8);
            float inv = 1.0f / rs;
            os0[e] += o0[e] * inv;
            os1[e] += o1[e] * inv;
        }
    }
#pragma unroll
    for (int e = 0; e < 8; ++e) {
        int tok = row0 + e + half * 8;
        _Float16* op = obuf + (b * NTOK + tok) * DD + h * HDIM;
        op[ln]      = (_Float16)os0[e];
        op[16 + ln] = (_Float16)os1[e];
    }
}

// ---------------------------------------------------------------------------
// 5) bilinear x4 upsample (align_corners) from [8192 tokens][512] f32 map
// ---------------------------------------------------------------------------
__global__ __launch_bounds__(256) void upsample_kernel(
    const float* __restrict__ sm, float* __restrict__ out) {
    long idx = (long)blockIdx.x * 256 + threadIdx.x;     // 8*512*128*128
    int xx = (int)(idx & 127);
    int yy = (int)((idx >> 7) & 127);
    int o  = (int)((idx >> 14) & 511);
    int b  = (int)(idx >> 23);
    const float s = 31.0f / 127.0f;
    float py = yy * s; int y0 = (int)py; y0 = y0 > 30 ? 30 : y0; float fy = py - y0;
    float px = xx * s; int x0 = (int)px; x0 = x0 > 30 ? 30 : x0; float fx = px - x0;
    int base = b * NTOK;
    const float* r0 = sm + ((base + (y0    ) * 32 + x0) << 9) + o;
    const float* r1 = sm + ((base + (y0 + 1) * 32 + x0) << 9) + o;
    float v0 = r0[0] * (1.0f - fx) + r0[512] * fx;
    float v1 = r1[0] * (1.0f - fx) + r1[512] * fx;
    out[idx] = v0 * (1.0f - fy) + v1 * fy;
}

// ---------------------------------------------------------------------------
// launcher
// ---------------------------------------------------------------------------
extern "C" void kernel_launch(void* const* d_in, const int* in_sizes, int n_in,
                              void* d_out, int out_size, void* d_ws, size_t ws_size,
                              hipStream_t stream) {
    const float* x       = (const float*)d_in[0];
    const float* le_w    = (const float*)d_in[1];
    const float* le_b    = (const float*)d_in[2];
    const float* mx_w    = (const float*)d_in[3];
    const float* mx_b    = (const float*)d_in[4];
    const float* av_w    = (const float*)d_in[5];
    const float* av_b    = (const float*)d_in[6];
    const float* le_g    = (const float*)d_in[7];
    const float* le_beta = (const float*)d_in[8];
    const float* le_m    = (const float*)d_in[9];
    const float* le_v    = (const float*)d_in[10];
    const float* mx_g    = (const float*)d_in[11];
    const float* mx_beta = (const float*)d_in[12];
    const float* mx_m    = (const float*)d_in[13];
    const float* mx_v    = (const float*)d_in[14];
    const float* av_g    = (const float*)d_in[15];
    const float* av_beta = (const float*)d_in[16];
    const float* av_m    = (const float*)d_in[17];
    const float* av_v    = (const float*)d_in[18];
    const float* q_w     = (const float*)d_in[19];
    const float* kv_w    = (const float*)d_in[20];
    const float* proj_w  = (const float*)d_in[21];
    const float* proj_b  = (const float*)d_in[22];
    const float* rpb     = (const float*)d_in[23];
    const float* co_w    = (const float*)d_in[24];
    const float* co_b    = (const float*)d_in[25];

    char* ws = (char*)d_ws;
    const size_t MB = 1u << 20;
    _Float16* tq   = (_Float16*)(ws + 0 * MB);      // 2 MB each
    _Float16* tm_  = (_Float16*)(ws + 2 * MB);
    _Float16* ta_  = (_Float16*)(ws + 4 * MB);
    _Float16* qf   = (_Float16*)(ws + 6 * MB);
    _Float16* kv1  = (_Float16*)(ws + 8 * MB);      // 4 MB
    _Float16* kv2  = (_Float16*)(ws + 12 * MB);     // 4 MB
    _Float16* vt1  = (_Float16*)(ws + 16 * MB);
    _Float16* vt2  = (_Float16*)(ws + 18 * MB);
    _Float16* ob   = (_Float16*)(ws + 20 * MB);
    _Float16* pj   = (_Float16*)(ws + 22 * MB);
    float*    co   = (float*)   (ws + 24 * MB);     // 16 MB
    _Float16* qwT  = (_Float16*)(ws + 40 * MB);                 // 32 KB
    _Float16* kvT  = (_Float16*)(ws + 40 * MB + (32u << 10));   // 64 KB
    _Float16* pwT  = (_Float16*)(ws + 40 * MB + (96u << 10));   // 32 KB
    _Float16* cowT = (_Float16*)(ws + 40 * MB + (128u << 10));  // 128 KB

    // 0) weights
    prep_kernel<<<256, 256, 0, stream>>>(q_w, kv_w, proj_w, co_w, qwT, kvT, pwT, cowT);

    // 1) front-end (reads x once)
    FrontArgs fa{x, le_w, le_b, le_g, le_beta, le_m, le_v,
                 mx_w, mx_b, mx_g, mx_beta, mx_m, mx_v,
                 av_w, av_b, av_g, av_beta, av_m, av_v,
                 tq, tm_, ta_};
    frontend_kernel<<<NB * DD * 4, 256, 0, stream>>>(fa);

    // 2) projections (WMMA GEMMs). q scale = 1/sqrt(32)
    //    grid = 512 M-tiles * (nout/32) N-pairs / 4 waves
    gemm_wmma<true,  false><<<512,  128, 0, stream>>>(tq,  qwT, nullptr, 0.0f, 0.1767766953f, 128, qf);
    gemm_wmma<true,  false><<<1024, 128, 0, stream>>>(tm_, kvT, nullptr, 0.0f, 1.0f,          256, kv1);
    gemm_wmma<true,  false><<<1024, 128, 0, stream>>>(ta_, kvT, nullptr, 0.0f, 1.0f,          256, kv2);

    // 3) V transposes
    vt_kernel<<<4096, 256, 0, stream>>>(kv1, kv2, vt1, vt2);

    // 4) attention (both branches, summed into ob)
    attn_kernel<<<dim3(16, NHEAD, NB), 128, 0, stream>>>(qf, kv1, kv2, vt1, vt2, rpb, ob);

    // 5) proj (bias x2 because both branches add proj_b), then channel-out GEMM
    gemm_wmma<true,  true><<<512,  128, 0, stream>>>(ob, pwT, proj_b, 2.0f, 1.0f, 128, pj);
    gemm_wmma<false, true><<<2048, 128, 0, stream>>>(pj, cowT, co_b,  1.0f, 1.0f, 512, co);

    // 6) bilinear x4 upsample -> d_out
    upsample_kernel<<<262144, 256, 0, stream>>>(co, (float*)d_out);
}
